// SchNetEnergy_75479755260178
// MI455X (gfx1250) — matrix operand order repaired and checked by reading, not compiled, not run on hardware
//
#include <hip/hip_runtime.h>

#define HID 128
#define NRBF 64
#define NBLK 4
#define NNODE 8192      // B*N
#define NEDGE 262144
#define NGRAPH 64
#define RCUTF 6.0f
#define GAMMAF (10.0f/36.0f)

typedef __attribute__((ext_vector_type(16))) __bf16        v16bf;
typedef __attribute__((ext_vector_type(2)))  __bf16        v2bf;
typedef __attribute__((ext_vector_type(8)))  float         v8f;
typedef __attribute__((ext_vector_type(8)))  unsigned int  v8u;
typedef __attribute__((ext_vector_type(4)))  unsigned int  v4u;

union FragU { v16bf v; unsigned short s[16]; v2bf h[8]; __bf16 b[16]; };
struct V4Pair { v4u a, b; };

// Native f32 -> bf16 conversion (RNE); backend may select v_cvt_pk_bf16_f32.
static __device__ __forceinline__ __bf16 tobf(float f) { return (__bf16)f; }
static __device__ __forceinline__ unsigned short bfbits(float f) {
  return __builtin_bit_cast(unsigned short, (__bf16)f);
}

static __device__ __forceinline__ float siluf(float x) {
  return x * (1.0f / (1.0f + __expf(-x)));
}
static __device__ __forceinline__ v8f wmma_bf16(v16bf a, v16bf b, v8f c) {
  return __builtin_amdgcn_wmma_f32_16x16x32_bf16(false, a, false, b, (short)0, c, false, false);
}

// ---------------------------------------------------------------- utilities
__global__ void fill_kernel(float* __restrict__ p, float v, int n) {
  int i = blockIdx.x * blockDim.x + threadIdx.x;
  if (i < n) p[i] = v;
}

__global__ void embed_kernel(const int* __restrict__ Z, const float* __restrict__ W,
                             float* __restrict__ h) {
  int i = blockIdx.x * 256 + threadIdx.x;           // NNODE*HID threads, exact
  int nidx = i >> 7, c = i & 127;
  h[i] = W[(size_t)Z[nidx] * HID + c];
}

__global__ void dist_kernel(const int* __restrict__ src, const int* __restrict__ dst,
                            const float* __restrict__ pos, float* __restrict__ d) {
  int i = blockIdx.x * 256 + threadIdx.x;           // NEDGE exact
  int s = src[i], t = dst[i];
  float dx = pos[s*3+0] - pos[t*3+0];
  float dy = pos[s*3+1] - pos[t*3+1];
  float dz = pos[s*3+2] - pos[t*3+2];
  d[i] = fminf(sqrtf(dx*dx + dy*dy + dz*dz), RCUTF);
}

// Pre-swizzle a row-major [K,128] f32 weight into bf16 WMMA B-fragment layout:
// frag index f = kc*8+nt ; element e of lane l holds W[kc*32 + (l>=16?8:0) + (e&7) + (e>=8?16:0)][nt*16 + l%16]
__global__ void swz_kernel(const float* __restrict__ W, unsigned short* __restrict__ o, int K) {
  int idx = blockIdx.x * 256 + threadIdx.x;
  if (idx >= K * HID) return;
  int e  = idx & 15;
  int l  = (idx >> 4) & 31;
  int f  = idx >> 9;
  int kc = f >> 3, nt = f & 7;
  int k  = kc * 32 + ((l >> 4) & 1) * 8 + (e & 7) + (e >> 3) * 16;
  int n  = nt * 16 + (l & 15);
  o[idx] = bfbits(W[k * HID + n]);
}

// ------------------------------------------------- generic 8192x128x128 GEMM
// out = act(A @ Wswz + bias) ; A f32 row-major, Wswz pre-swizzled bf16.
// 512 waves, one 16-row M-tile per wave, all 8 N-tiles.
__global__ void gemm128_kernel(const float* __restrict__ A, const unsigned short* __restrict__ Wswz,
                               const float* __restrict__ bias, float* __restrict__ out, int act) {
  int lane = threadIdx.x & 31;
  int gw   = blockIdx.x * 8 + (threadIdx.x >> 5);   // 0..511
  int lm   = lane & 15, hi = lane >> 4;
  int m0   = gw * 16;

  v16bf afrag[4];
#pragma unroll
  for (int kc = 0; kc < 4; ++kc) {
    const float* ap = A + (size_t)(m0 + lm) * HID + kc * 32 + hi * 8;
    v8f a0 = *(const v8f*)ap;          // K offs +0..7
    v8f a1 = *(const v8f*)(ap + 16);   // K offs +16..23
    FragU f;
#pragma unroll
    for (int e = 0; e < 8; ++e) { f.b[e] = tobf(a0[e]); f.b[8 + e] = tobf(a1[e]); }
    afrag[kc] = f.v;
  }

  v8f c[8] = {};
#pragma unroll
  for (int nt = 0; nt < 8; ++nt)
#pragma unroll
    for (int kc = 0; kc < 4; ++kc) {
      v16bf b = __builtin_bit_cast(v16bf, *(const v8u*)(Wswz + ((kc * 8 + nt) * 32 + lane) * 16));
      c[nt] = wmma_bf16(afrag[kc], b, c[nt]);
    }

#pragma unroll
  for (int nt = 0; nt < 8; ++nt) {
    float bb = bias ? bias[nt * 16 + lm] : 0.0f;
#pragma unroll
    for (int r = 0; r < 8; ++r) {
      float v = c[nt][r] + bb;
      if (act) v = siluf(v);
      out[(size_t)(m0 + r + hi * 8) * HID + nt * 16 + lm] = v;
    }
  }
}

// ------------------------------------------------------- fused edge pipeline
// Per 16-edge tile (one wave): rbf -> GEMM1(16x64x128) -> SiLU -> LDS transpose
// -> GEMM2(16x128x128) -> (+b2) * gather(t[src]) -> atomic scatter to agg[dst]
__global__ void edge_kernel(const float* __restrict__ dvec, const int* __restrict__ src,
                            const int* __restrict__ dst, const float* __restrict__ t,
                            const unsigned short* __restrict__ w1swz,
                            const unsigned short* __restrict__ w2swz,
                            const float* __restrict__ b1, const float* __restrict__ b2,
                            float* __restrict__ agg) {
  __shared__ __align__(16) unsigned short xb[8 * 16 * HID];   // 32 KB, per-wave regions
  int lane = threadIdx.x & 31;
  int wv   = threadIdx.x >> 5;
  int gw   = blockIdx.x * 8 + wv;                 // 0..2047
  int lm   = lane & 15, hi = lane >> 4;
  unsigned short* xw = xb + wv * (16 * HID);

  float bb1[8], bb2[8];
#pragma unroll
  for (int nt = 0; nt < 8; ++nt) { bb1[nt] = b1[nt * 16 + lm]; bb2[nt] = b2[nt * 16 + lm]; }

  for (int it = 0; it < 8; ++it) {                // 2048 waves x 8 = 16384 tiles exact
    int tile = gw + it * 2048;
    int eid  = tile * 16 + lm;
    float dv = dvec[eid];
    int sid  = src[eid], did = dst[eid];

    // rbf computed directly into A-fragment layout (16x32 bf16 per chunk)
    v16bf a1f[2];
#pragma unroll
    for (int kc = 0; kc < 2; ++kc) {
      FragU f;
#pragma unroll
      for (int e = 0; e < 16; ++e) {
        int k    = kc * 32 + hi * 8 + e + ((e >= 8) ? 8 : 0);
        float ck = (float)k * (RCUTF / 63.0f);
        float dd = dv - ck;
        f.b[e]   = tobf(__expf(-GAMMAF * dd * dd));
      }
      a1f[kc] = f.v;
    }

    // GEMM1: [16,64] @ [64,128]
    v8f c1[8] = {};
#pragma unroll
    for (int nt = 0; nt < 8; ++nt)
#pragma unroll
      for (int kc = 0; kc < 2; ++kc) {
        v16bf b = __builtin_bit_cast(v16bf, *(const v8u*)(w1swz + ((kc * 8 + nt) * 32 + lane) * 16));
        c1[nt] = wmma_bf16(a1f[kc], b, c1[nt]);
      }

    // bias + SiLU -> LDS row-major [16][128] bf16 (transpose bounce)
#pragma unroll
    for (int nt = 0; nt < 8; ++nt)
#pragma unroll
      for (int r = 0; r < 8; ++r) {
        float v = siluf(c1[nt][r] + bb1[nt]);
        xw[(r + hi * 8) * HID + nt * 16 + lm] = bfbits(v);
      }

    // GEMM2: [16,128] @ [128,128], A-fragments re-read from LDS
    v8f c2[8] = {};
#pragma unroll
    for (int kc = 0; kc < 4; ++kc) {
      V4Pair p;
      p.a = *(const v4u*)(xw + lm * HID + kc * 32 + hi * 8);
      p.b = *(const v4u*)(xw + lm * HID + kc * 32 + hi * 8 + 16);
      v16bf a2 = __builtin_bit_cast(v16bf, p);
#pragma unroll
      for (int nt = 0; nt < 8; ++nt) {
        v16bf b = __builtin_bit_cast(v16bf, *(const v8u*)(w2swz + ((kc * 8 + nt) * 32 + lane) * 16));
        c2[nt] = wmma_bf16(a2, b, c2[nt]);
      }
    }

    // epilogue: m = (w + b2) * t[src]  -> atomic add into agg[dst]
#pragma unroll
    for (int r = 0; r < 8; ++r) {
      int mm = r + hi * 8;
      int s2 = __shfl(sid, mm, 32);
      int d2 = __shfl(did, mm, 32);
      const float* trow = t + (size_t)s2 * HID;
      float* arow       = agg + (size_t)d2 * HID;
#pragma unroll
      for (int nt = 0; nt < 8; ++nt) {
        float v = (c2[nt][r] + bb2[nt]) * trow[nt * 16 + lm];
        unsafeAtomicAdd(arow + nt * 16 + lm, v);
      }
    }
  }
}

// --------------------------------------------------- residual + LayerNorm
__global__ void ln_kernel(float* __restrict__ h, const float* __restrict__ o,
                          const float* __restrict__ g, const float* __restrict__ b) {
  int lane = threadIdx.x & 31;
  int row  = blockIdx.x * 8 + (threadIdx.x >> 5);   // 0..8191
  size_t base = (size_t)row * HID;
  float x[4]; float s = 0.f;
#pragma unroll
  for (int j = 0; j < 4; ++j) { x[j] = h[base + lane + j * 32] + o[base + lane + j * 32]; s += x[j]; }
#pragma unroll
  for (int off = 16; off > 0; off >>= 1) s += __shfl_xor(s, off, 32);
  float mu = s * (1.0f / HID);
  float vv = 0.f;
#pragma unroll
  for (int j = 0; j < 4; ++j) { float dd = x[j] - mu; vv += dd * dd; }
#pragma unroll
  for (int off = 16; off > 0; off >>= 1) vv += __shfl_xor(vv, off, 32);
  float inv = rsqrtf(vv * (1.0f / HID) + 1e-5f);
#pragma unroll
  for (int j = 0; j < 4; ++j) {
    int cc = lane + j * 32;
    h[base + cc] = (x[j] - mu) * inv * g[cc] + b[cc];
  }
}

// ------------------------------------------------------------- readout
__global__ void readout_kernel(const float* __restrict__ h, const float* __restrict__ w1,
                               const float* __restrict__ b1, const float* __restrict__ w2,
                               const float* __restrict__ b2, float* __restrict__ out) {
  __shared__ float sh[8][HID];
  int lane = threadIdx.x & 31;
  int wv   = threadIdx.x >> 5;
  int atom = blockIdx.x * 8 + wv;                   // 0..8191
#pragma unroll
  for (int j = 0; j < 4; ++j) {
    int cc = lane + j * 32;
    sh[wv][cc] = siluf(h[(size_t)atom * HID + cc]);
  }
  float x0 = b1[lane], x1 = b1[lane + 32];
  for (int k = 0; k < HID; ++k) {
    float hv = sh[wv][k];
    x0 += hv * w1[k * 64 + lane];
    x1 += hv * w1[k * 64 + lane + 32];
  }
  float p = siluf(x0) * w2[lane] + siluf(x1) * w2[lane + 32];
#pragma unroll
  for (int off = 16; off > 0; off >>= 1) p += __shfl_xor(p, off, 32);
  if (lane == 0) unsafeAtomicAdd(&out[atom >> 7], p + b2[0]);
}

// ---------------------------------------------------------------- launcher
extern "C" void kernel_launch(void* const* d_in, const int* in_sizes, int n_in,
                              void* d_out, int out_size, void* d_ws, size_t ws_size,
                              hipStream_t stream) {
  (void)in_sizes; (void)n_in; (void)out_size; (void)ws_size;
  const int*   Z    = (const int*)d_in[0];
  const float* pos  = (const float*)d_in[1];
  const int*   eidx = (const int*)d_in[2];
  const float* embw = (const float*)d_in[3];
  const float* ew1  = (const float*)d_in[4];
  const float* eb1  = (const float*)d_in[5];
  const float* ew2  = (const float*)d_in[6];
  const float* eb2  = (const float*)d_in[7];
  const float* lin  = (const float*)d_in[8];
  const float* nw1  = (const float*)d_in[9];
  const float* nb1  = (const float*)d_in[10];
  const float* nw2  = (const float*)d_in[11];
  const float* nb2  = (const float*)d_in[12];
  const float* lng  = (const float*)d_in[13];
  const float* lnb  = (const float*)d_in[14];
  const float* rw1  = (const float*)d_in[15];
  const float* rb1  = (const float*)d_in[16];
  const float* rw2  = (const float*)d_in[17];
  const float* rb2  = (const float*)d_in[18];
  float* out = (float*)d_out;

  char* ws = (char*)d_ws;
  const size_t NMATB = (size_t)NNODE * HID * 4;     // bytes per [8192,128] f32
  float* h    = (float*)(ws);
  float* tb   = (float*)(ws + NMATB);
  float* agg  = (float*)(ws + NMATB * 2);
  float* ub   = (float*)(ws + NMATB * 3);
  float* ob   = (float*)(ws + NMATB * 4);
  float* dist = (float*)(ws + NMATB * 5);
  unsigned short* slin = (unsigned short*)(ws + NMATB * 5 + (size_t)NEDGE * 4);
  unsigned short* se1  = slin + 4 * 16384;
  unsigned short* se2  = se1  + 4 * 8192;
  unsigned short* sn1  = se2  + 4 * 16384;
  unsigned short* sn2  = sn1  + 4 * 16384;

  const int* esrc = eidx;
  const int* edst = eidx + NEDGE;

  for (int i = 0; i < NBLK; ++i) {
    swz_kernel<<<64, 256, 0, stream>>>(lin + (size_t)i * HID * HID,  slin + i * 16384, 128);
    swz_kernel<<<32, 256, 0, stream>>>(ew1 + (size_t)i * NRBF * HID, se1  + i * 8192,  64);
    swz_kernel<<<64, 256, 0, stream>>>(ew2 + (size_t)i * HID * HID,  se2  + i * 16384, 128);
    swz_kernel<<<64, 256, 0, stream>>>(nw1 + (size_t)i * HID * HID,  sn1  + i * 16384, 128);
    swz_kernel<<<64, 256, 0, stream>>>(nw2 + (size_t)i * HID * HID,  sn2  + i * 16384, 128);
  }
  embed_kernel<<<4096, 256, 0, stream>>>(Z, embw, h);
  dist_kernel<<<1024, 256, 0, stream>>>(esrc, edst, pos, dist);

  for (int i = 0; i < NBLK; ++i) {
    fill_kernel<<<4096, 256, 0, stream>>>(agg, 0.0f, NNODE * HID);
    gemm128_kernel<<<64, 256, 0, stream>>>(h, slin + i * 16384, nullptr, tb, 0);
    edge_kernel<<<256, 256, 0, stream>>>(dist, esrc, edst, tb, se1 + i * 8192, se2 + i * 16384,
                                         eb1 + i * HID, eb2 + i * HID, agg);
    gemm128_kernel<<<64, 256, 0, stream>>>(agg, sn1 + i * 16384, nb1 + i * HID, ub, 1);
    gemm128_kernel<<<64, 256, 0, stream>>>(ub,  sn2 + i * 16384, nb2 + i * HID, ob, 0);
    ln_kernel<<<1024, 256, 0, stream>>>(h, ob, lng + i * HID, lnb + i * HID);
  }
  fill_kernel<<<1, 64, 0, stream>>>(out, 0.0f, NGRAPH);
  readout_kernel<<<1024, 256, 0, stream>>>(h, rw1, rb1, rw2, rb2, out);
}
